// Solograph_HeteroGNN_79456894976243
// MI455X (gfx1250) — matrix operand order repaired and checked by reading, not compiled, not run on hardware
//
#include <hip/hip_runtime.h>

// ---------------------------------------------------------------------------
// Solograph HeteroGNN (2-layer GATv2 heteroconv) for MI455X / gfx1250.
// Dense matmuls via v_wmma_f32_16x16x32_bf16 (bf16 in, f32 accumulate).
//   - 32xK A-tile staged once in LDS (bf16), K-loop runs barrier-free.
//   - each wave owns two 16x16 accumulators (B fragment reused x2).
// Edge phase: wave-per-edge gather/scatter, f32 global atomics (L2 resident).
// ---------------------------------------------------------------------------

typedef __attribute__((ext_vector_type(16))) __bf16 v16bf;
typedef __attribute__((ext_vector_type(8)))  __bf16 v8bf;
typedef __attribute__((ext_vector_type(8)))  float  v8f;

#define HID   256
#define NMOD  12800
#define NB    128
#define LSEQ  100
#define E_MM  160000
#define E_MZ  12800

// ---------------- utility fills ----------------
__global__ void k_fill_u32(unsigned* __restrict__ p, unsigned v, long n) {
  long i = (long)blockIdx.x * blockDim.x + threadIdx.x;
  if (i < n) p[i] = v;
}

// ---------------- weight transpose + f32->bf16: W[k][n] -> Wt[n][k] ----------------
__global__ void k_wT_bf16(const float* __restrict__ W, __bf16* __restrict__ Wt,
                          int K, int N, long nmat) {
  long per = (long)K * N;
  long idx = (long)blockIdx.x * blockDim.x + threadIdx.x;
  if (idx >= per * nmat) return;
  long m = idx / per, r = idx - m * per;
  int n = (int)(r / K), k = (int)(r - (long)n * K);
  Wt[m * per + (long)n * K + k] = (__bf16)W[m * per + (long)k * N + n];
}

// ---------------- bf16 WMMA GEMM ----------------
// C[M,N=256] = A[M,K] (f32, converted on the fly) @ B[K,256] (+ bias[col]).
// B supplied transposed as Bt[256,K] bf16 so each lane's fragment is contiguous.
// Block: 256 thr = 8 waves; block tile M=32 x N=128; grid = (M/32, N/128).
// Dynamic LDS: 32*K bf16 (<= 48 KB for K=768). K % 32 == 0, M % 32 == 0.
// Fragment layouts per CDNA5 ISA 7.12.2 (wave32):
//   A 16x32: lane half 0 holds K 0-7 & 16-23, half 1 holds K 8-15 & 24-31.
//   B 32x16: lane half 0 holds K 0-7 & 16-23, half 1 holds K 8-15 & 24-31 (col = lane&15).
//   C: VGPR v -> row 8*(lane>>4)+v, col lane&15.
__global__ __launch_bounds__(256)
void k_gemm_bf16(const float* __restrict__ A, const __bf16* __restrict__ Bt,
                 const float* __restrict__ bias, float* __restrict__ C,
                 int M, int N, int K) {
  extern __shared__ __bf16 lds_a[];  // [32][K]
  const int tid  = threadIdx.x;
  const int wave = tid >> 5, lane = tid & 31;
  const int half = lane >> 4, lcol = lane & 15;
  const int mBase = blockIdx.x * 32;
  const int nBase = blockIdx.y * 128 + wave * 16;

  // stage the entire 32 x K A-tile as bf16 into LDS (shared by all 8 waves)
  for (int e = tid * 2; e < 32 * K; e += 512) {
    int r = e / K, cc = e - r * K;
    float2 av = *(const float2*)(A + (size_t)(mBase + r) * K + cc);
    lds_a[e]     = (__bf16)av.x;
    lds_a[e + 1] = (__bf16)av.y;
  }
  __syncthreads();

  v8f c0, c1;
  float binit = bias ? bias[nBase + lcol] : 0.0f;
  #pragma unroll
  for (int i = 0; i < 8; ++i) { c0[i] = binit; c1[i] = binit; }

  const int koff = half * 8;
  for (int k0 = 0; k0 < K; k0 += 32) {
    // B fragment: col = nBase+lcol, K in [k0 + half*8, +8) and [k0 + 16 + half*8, +8)
    const __bf16* bp = Bt + (size_t)(nBase + lcol) * K + k0 + half * 8;
    v8bf blo = *(const v8bf*)bp;
    v8bf bhi = *(const v8bf*)(bp + 16);
    v16bf b;
    #pragma unroll
    for (int i = 0; i < 8; ++i) { b[i] = blo[i]; b[i + 8] = bhi[i]; }

    // A fragment for M-subtile 0 (rows 0-15)
    {
      const __bf16* ap = &lds_a[(size_t)lcol * K + k0 + koff];
      v8bf alo = *(const v8bf*)ap;
      v8bf ahi = *(const v8bf*)(ap + 16);
      v16bf a;
      #pragma unroll
      for (int i = 0; i < 8; ++i) { a[i] = alo[i]; a[i + 8] = ahi[i]; }
      c0 = __builtin_amdgcn_wmma_f32_16x16x32_bf16(false, a, false, b,
                                                   (short)0, c0, false, false);
    }
    // A fragment for M-subtile 1 (rows 16-31), reusing the same B fragment
    {
      const __bf16* ap = &lds_a[(size_t)(16 + lcol) * K + k0 + koff];
      v8bf alo = *(const v8bf*)ap;
      v8bf ahi = *(const v8bf*)(ap + 16);
      v16bf a;
      #pragma unroll
      for (int i = 0; i < 8; ++i) { a[i] = alo[i]; a[i + 8] = ahi[i]; }
      c1 = __builtin_amdgcn_wmma_f32_16x16x32_bf16(false, a, false, b,
                                                   (short)0, c1, false, false);
    }
  }

  #pragma unroll
  for (int v = 0; v < 8; ++v) {
    int row0 = mBase + half * 8 + v;
    C[(size_t)row0 * N + nBase + lcol] = c0[v];
    C[(size_t)(row0 + 16) * N + nBase + lcol] = c1[v];
  }
}

// ---------------- sinusoidal PE add (positions reset every LSEQ rows) ----------------
__global__ void k_pe_add(float* __restrict__ x, long n) {  // x: [N,256]
  long i = (long)blockIdx.x * blockDim.x + threadIdx.x;
  if (i >= n) return;
  int row = (int)(i >> 8), d = (int)(i & 255);
  float pos = (float)(row % LSEQ);
  int j2 = d & ~1;  // 2*j
  float inv = __expf(-logf(10000.0f) * (float)j2 * (1.0f / 256.0f));
  float ang = pos * inv;
  x[i] += (d & 1) ? __cosf(ang) : __sinf(ang);
}

// ---------------- float atomic max via signed/unsigned bit trick ----------------
__device__ __forceinline__ void atomicMaxF(float* addr, float v) {
  if (v >= 0.0f) atomicMax((int*)addr, __float_as_int(v));
  else           atomicMin((unsigned int*)addr, __float_as_uint(v));
}

// ---------------- edge pass 1: logits + segment max (wave per edge) ----------------
__global__ __launch_bounds__(256)
void k_edge_logits(const int* __restrict__ ei, int E,
                   const float* __restrict__ hl, const float* __restrict__ hr,
                   const float* __restrict__ att,      // [256] = [4 heads][64]
                   float* __restrict__ logits,         // [E,4]
                   float* __restrict__ mmax) {         // [Ndst,4], pre-filled -inf
  int wave = threadIdx.x >> 5, lane = threadIdx.x & 31;
  int e = blockIdx.x * 8 + wave;
  if (e >= E) return;
  int src = ei[e], dst = ei[E + e];
  int d0 = lane * 8;  // 8 consecutive dims per lane; head = lane>>3
  const float4* pl = (const float4*)(hl + (size_t)src * HID + d0);
  const float4* pr = (const float4*)(hr + (size_t)dst * HID + d0);
  const float4* pa = (const float4*)(att + d0);
  float part = 0.0f;
  #pragma unroll
  for (int q = 0; q < 2; ++q) {
    float4 l = pl[q], r = pr[q], a = pa[q];
    float s;
    s = l.x + r.x; part += a.x * (s > 0.0f ? s : 0.2f * s);
    s = l.y + r.y; part += a.y * (s > 0.0f ? s : 0.2f * s);
    s = l.z + r.z; part += a.z * (s > 0.0f ? s : 0.2f * s);
    s = l.w + r.w; part += a.w * (s > 0.0f ? s : 0.2f * s);
  }
  part += __shfl_xor(part, 1, 32);
  part += __shfl_xor(part, 2, 32);
  part += __shfl_xor(part, 4, 32);
  if ((lane & 7) == 0) {
    int h = lane >> 3;
    logits[(size_t)e * 4 + h] = part;
    atomicMaxF(&mmax[(size_t)dst * 4 + h], part);
  }
}

// ---------------- edge pass 2: exp + segment sum ----------------
__global__ void k_edge_exden(const int* __restrict__ ei, int E,
                             const float* __restrict__ logits,
                             const float* __restrict__ mmax,
                             float* __restrict__ ex, float* __restrict__ den) {
  long i = (long)blockIdx.x * blockDim.x + threadIdx.x;
  if (i >= (long)E * 4) return;
  int e = (int)(i >> 2), h = (int)(i & 3);
  int dst = ei[E + e];
  float m = mmax[(size_t)dst * 4 + h];
  if (m < -3.0e38f) m = 0.0f;  // reference: non-finite max -> 0
  float v = __expf(logits[i] - m);
  ex[i] = v;
  atomicAdd(&den[(size_t)dst * 4 + h], v);
}

// ---------------- edge pass 3: weighted scatter (wave per edge) ----------------
__global__ __launch_bounds__(256)
void k_edge_scatter(const int* __restrict__ ei, int E,
                    const float* __restrict__ hl,
                    const float* __restrict__ ex, const float* __restrict__ den,
                    float* __restrict__ acc) {
  int wave = threadIdx.x >> 5, lane = threadIdx.x & 31;
  int e = blockIdx.x * 8 + wave;
  if (e >= E) return;
  int src = ei[e], dst = ei[E + e];
  int h = lane >> 3;
  float alpha = ex[(size_t)e * 4 + h] / (den[(size_t)dst * 4 + h] + 1e-16f);
  int d0 = lane * 8;
  const float4* pl = (const float4*)(hl + (size_t)src * HID + d0);
  float* pa = acc + (size_t)dst * HID + d0;
  #pragma unroll
  for (int q = 0; q < 2; ++q) {
    float4 l = pl[q];
    atomicAdd(pa + q * 4 + 0, alpha * l.x);
    atomicAdd(pa + q * 4 + 1, alpha * l.y);
    atomicAdd(pa + q * 4 + 2, alpha * l.z);
    atomicAdd(pa + q * 4 + 3, alpha * l.w);
  }
}

// ---------------- per-node finalize: mean of 4 incoming types (+biases), relu --------
__global__ void k_finalize(const float* __restrict__ acc,
                           const float* __restrict__ biasL,  // [16][256] for layer l
                           int t0, int t1, int t2, int t3,
                           float* __restrict__ xd, long n) {
  long i = (long)blockIdx.x * blockDim.x + threadIdx.x;
  if (i >= n) return;
  int d = (int)(i & 255);
  float b = biasL[t0 * HID + d] + biasL[t1 * HID + d] +
            biasL[t2 * HID + d] + biasL[t3 * HID + d];
  float v = (acc[i] + b) * 0.25f;
  xd[i] = v > 0.0f ? v : 0.0f;
}

// ---------------- scene mean over the 3 modalities ----------------
__global__ void k_scene_mean(const float* __restrict__ xt,
                             const float* __restrict__ xa,
                             const float* __restrict__ xv,
                             float* __restrict__ out) {  // [128,256]
  int b = blockIdx.x, d = threadIdx.x;
  float s = 0.0f;
  for (int i = 0; i < LSEQ; ++i) {
    size_t r = (size_t)(b * LSEQ + i) * HID + d;
    s += xt[r] + xa[r] + xv[r];
  }
  out[(size_t)b * HID + d] = s * (1.0f / (3.0f * LSEQ));
}

// ---------------------------------------------------------------------------
extern "C" void kernel_launch(void* const* d_in, const int* in_sizes, int n_in,
                              void* d_out, int out_size, void* d_ws, size_t ws_size,
                              hipStream_t stream) {
  const float* x_text  = (const float*)d_in[0];
  const float* x_audio = (const float*)d_in[1];
  const float* x_video = (const float*)d_in[2];
  const float* x_z     = (const float*)d_in[3];
  const float* W_text  = (const float*)d_in[4];
  const float* b_text  = (const float*)d_in[5];
  const float* W_audio = (const float*)d_in[6];
  const float* b_audio = (const float*)d_in[7];
  const float* W_video = (const float*)d_in[8];
  const float* b_video = (const float*)d_in[9];
  const float* Wl      = (const float*)d_in[10];
  const float* Wr      = (const float*)d_in[11];
  const float* att     = (const float*)d_in[12];
  const float* bias    = (const float*)d_in[13];
  const int*   ei_mm   = (const int*)d_in[18];
  const int*   ei_mz   = (const int*)d_in[19];
  const int*   ei_zz   = (const int*)d_in[20];
  float* out = (float*)d_out;

  // ---- workspace carve (≈120 MB) ----
  size_t off = 0;
  auto alloc = [&](size_t bytes) -> void* {
    void* p = (char*)d_ws + off;
    off += (bytes + 255) & ~(size_t)255;
    return p;
  };
  __bf16* WlT  = (__bf16*)alloc((size_t)32 * 256 * 256 * 2);
  __bf16* WrT  = (__bf16*)alloc((size_t)32 * 256 * 256 * 2);
  __bf16* WtTt = (__bf16*)alloc((size_t)256 * 768 * 2);
  __bf16* WtTa = (__bf16*)alloc((size_t)256 * 128 * 2);
  __bf16* WtTv = (__bf16*)alloc((size_t)256 * 512 * 2);
  float* xd_t = (float*)alloc((size_t)NMOD * HID * 4);
  float* xd_a = (float*)alloc((size_t)NMOD * HID * 4);
  float* xd_v = (float*)alloc((size_t)NMOD * HID * 4);
  float* xd_z = (float*)alloc((size_t)NB * HID * 4);
  float* hl     = (float*)alloc((size_t)NMOD * HID * 4);
  float* hr     = (float*)alloc((size_t)NMOD * HID * 4);
  float* logits = (float*)alloc((size_t)E_MM * 4 * 4);
  float* ex     = (float*)alloc((size_t)E_MM * 4 * 4);
  float* mbuf   = (float*)alloc((size_t)NMOD * 4 * 4);
  float* den    = (float*)alloc((size_t)NMOD * 4 * 4);
  float* acc    = (float*)alloc((size_t)(3 * NMOD + NB) * HID * 4);

  float* xdP[4]  = {xd_t, xd_a, xd_v, xd_z};
  float* accP[4] = {acc, acc + (size_t)NMOD * HID, acc + (size_t)2 * NMOD * HID,
                    acc + (size_t)3 * NMOD * HID};
  const int Nn[4] = {NMOD, NMOD, NMOD, NB};

  // edge-type metadata (ET order from the reference; types 0=t,1=a,2=v,3=z)
  const int srcT[16] = {0,0,0, 1,1,1, 2,2,2, 0,3, 1,3, 2,3, 3};
  const int dstT[16] = {0,1,2, 0,1,2, 0,1,2, 3,0, 3,1, 3,2, 3};
  const int* eiP[16]; int Ecnt[16];
  for (int t = 0; t < 9; ++t)  { eiP[t] = ei_mm + (size_t)t * 2 * E_MM; Ecnt[t] = E_MM; }
  for (int t = 9; t < 15; ++t) { eiP[t] = ei_mz + (size_t)(t - 9) * 2 * E_MZ; Ecnt[t] = E_MZ; }
  eiP[15] = ei_zz; Ecnt[15] = NB;
  const int inc[4][4] = {{0,3,6,10}, {1,4,7,12}, {2,5,8,14}, {9,11,13,15}};

  auto fill = [&](void* p, unsigned v, long n) {
    k_fill_u32<<<dim3((unsigned)((n + 255) / 256)), 256, 0, stream>>>((unsigned*)p, v, n);
  };
  auto gemm = [&](const float* A, const __bf16* Bt, const float* bi, float* C,
                  int M, int K) {
    size_t smem = (size_t)32 * K * 2;  // 32 x K bf16 A-tile
    k_gemm_bf16<<<dim3(M / 32, HID / 128), 256, smem, stream>>>(A, Bt, bi, C, M, HID, K);
  };

  // ---- weight prep: transpose + bf16 ----
  k_wT_bf16<<<dim3((768u * 256 + 255) / 256), 256, 0, stream>>>(W_text,  WtTt, 768, 256, 1);
  k_wT_bf16<<<dim3((128u * 256 + 255) / 256), 256, 0, stream>>>(W_audio, WtTa, 128, 256, 1);
  k_wT_bf16<<<dim3((512u * 256 + 255) / 256), 256, 0, stream>>>(W_video, WtTv, 512, 256, 1);
  k_wT_bf16<<<dim3((32u * 65536 + 255) / 256), 256, 0, stream>>>(Wl, WlT, 256, 256, 32);
  k_wT_bf16<<<dim3((32u * 65536 + 255) / 256), 256, 0, stream>>>(Wr, WrT, 256, 256, 32);

  // ---- input projections (+PE for modalities; z reuses text proj, no PE) ----
  gemm(x_text,  WtTt, b_text,  xd_t, NMOD, 768);
  gemm(x_audio, WtTa, b_audio, xd_a, NMOD, 128);
  gemm(x_video, WtTv, b_video, xd_v, NMOD, 512);
  gemm(x_z,     WtTt, b_text,  xd_z, NB,   768);
  long nmodf = (long)NMOD * HID;
  k_pe_add<<<dim3((unsigned)((nmodf + 255) / 256)), 256, 0, stream>>>(xd_t, nmodf);
  k_pe_add<<<dim3((unsigned)((nmodf + 255) / 256)), 256, 0, stream>>>(xd_a, nmodf);
  k_pe_add<<<dim3((unsigned)((nmodf + 255) / 256)), 256, 0, stream>>>(xd_v, nmodf);

  // ---- 2 GATv2 heteroconv layers ----
  for (int l = 0; l < 2; ++l) {
    fill(acc, 0u, (long)(3 * NMOD + NB) * HID);
    for (int t = 0; t < 16; ++t) {
      const int s = srcT[t], d = dstT[t], E = Ecnt[t];
      const size_t wOff = ((size_t)l * 16 + t) * 256 * 256;
      gemm(xdP[s], WlT + wOff, nullptr, hl, Nn[s], HID);
      gemm(xdP[d], WrT + wOff, nullptr, hr, Nn[d], HID);
      fill(mbuf, 0xFF800000u, (long)Nn[d] * 4);  // -inf
      fill(den,  0u,          (long)Nn[d] * 4);
      const float* attP = att + ((size_t)l * 16 + t) * 256;
      k_edge_logits<<<dim3((E + 7) / 8), 256, 0, stream>>>(eiP[t], E, hl, hr, attP,
                                                           logits, mbuf);
      k_edge_exden<<<dim3(((unsigned)E * 4 + 255) / 256), 256, 0, stream>>>(
          eiP[t], E, logits, mbuf, ex, den);
      k_edge_scatter<<<dim3((E + 7) / 8), 256, 0, stream>>>(eiP[t], E, hl, ex, den,
                                                            accP[d]);
    }
    const float* biasL = bias + (size_t)l * 16 * 256;
    for (int nt = 0; nt < 4; ++nt) {
      long n = (long)Nn[nt] * HID;
      k_finalize<<<dim3((unsigned)((n + 255) / 256)), 256, 0, stream>>>(
          accP[nt], biasL, inc[nt][0], inc[nt][1], inc[nt][2], inc[nt][3],
          xdP[nt], n);
    }
  }

  // ---- scene mean over modality nodes ----
  k_scene_mean<<<dim3(NB), 256, 0, stream>>>(xd_t, xd_a, xd_v, out);
}
